// GPTLanguageModel_83425444757945
// MI455X (gfx1250) — matrix operand order repaired and checked by reading, not compiled, not run on hardware
//
#include <hip/hip_runtime.h>
#include <hip/hip_bf16.h>
#include <math.h>

// GPT forward for MI455X (gfx1250): bf16 WMMA GEMMs fed by the Tensor Data
// Mover (TENSOR_LOAD_TO_LDS, TENSORcnt-synchronized, double-buffered LDS).
#define B_  2
#define T_  2048
#define V_  32000
#define E_  1024
#define H_  16
#define L_  8
#define HS_ 64
#define FF_ 4096
#define BT_ (B_*T_)

#define HAVE_TDM __has_builtin(__builtin_amdgcn_tensor_load_to_lds)

typedef __attribute__((ext_vector_type(16))) __bf16 v16bf;
typedef __attribute__((ext_vector_type(8)))  __bf16 v8bf;
typedef __attribute__((ext_vector_type(8)))  float  v8f;
typedef __attribute__((ext_vector_type(4)))  unsigned int v4u;
typedef __attribute__((ext_vector_type(8)))  int v8i;
typedef __attribute__((ext_vector_type(4)))  int v4i;

static __device__ __forceinline__ __bf16 f2bf(float f) {
  unsigned u = __builtin_bit_cast(unsigned, f);
  u = (u + 0x7FFFu + ((u >> 16) & 1u)) >> 16;        // round-to-nearest-even
  unsigned short s = (unsigned short)u;
  return __builtin_bit_cast(__bf16, s);
}
static __device__ __forceinline__ float bf2f(__bf16 b) {
  unsigned short s = __builtin_bit_cast(unsigned short, b);
  unsigned u = ((unsigned)s) << 16;
  return __builtin_bit_cast(float, u);
}

// A-fragment (16x32 MxK, 16-bit) per ISA 7.12.2: m = lane%16,
// K = 8*(lane/16)+j (j<8), K = 16+8*(lane/16)+(j-8) (j>=8).
static __device__ __forceinline__ v16bf ldA(const __bf16* p_lo, const __bf16* p_hi) {
  v8bf lo = *(const v8bf*)p_lo;
  v8bf hi = *(const v8bf*)p_hi;
  v16bf r;
#pragma unroll
  for (int i = 0; i < 8; ++i) { r[i] = lo[i]; r[i + 8] = hi[i]; }
  return r;
}

// ---------------------------------------------------------------------------
// TDM: issue one 2-D tile copy (global bf16 -> LDS), per ISA 8.3/8.4 D#.
// tile_w elems contiguous per row, tile_h rows, row stride `stride` elems.
// Packs rows densely into LDS starting at lds_off (tile_w == LDS row stride).
// ---------------------------------------------------------------------------
static __device__ __forceinline__ void tdm_load_2d(
    unsigned lds_off, const __bf16* g, int tile_w, int tile_h, int stride)
{
#if HAVE_TDM
  unsigned long long ga = (unsigned long long)(size_t)g;
  v4u g0;
  g0[0] = 1u;                                            // count=1 (user D#)
  g0[1] = lds_off;                                       // lds_addr
  g0[2] = (unsigned)(ga & 0xFFFFFFFFu);                  // global_addr[31:0]
  g0[3] = (unsigned)((ga >> 32) & 0x01FFFFFFu) | (2u << 30); // [56:32] | type=2
  v8i g1;
  g1[0] = (int)(1u << 16);                               // data_size=2B
  g1[1] = 0;                                             // no atomic barrier; dim0 lo16=0
  g1[2] = 0x0100;                                        // tensor_dim0 = 1<<24
  g1[3] = (int)(0x0100u | ((unsigned)tile_w << 16));     // tensor_dim1=1<<24, tile_dim0
  g1[4] = tile_h;                                        // tile_dim1, tile_dim2=0
  g1[5] = stride;                                        // tensor_dim0_stride[31:0]
  g1[6] = 0; g1[7] = 0;
  v4i g2 = {0, 0, 0, 0};
  v4i g3 = {0, 0, 0, 0};
#if __clang_major__ >= 23
  v8i gx = {0, 0, 0, 0, 0, 0, 0, 0};
  __builtin_amdgcn_tensor_load_to_lds(g0, g1, g2, g3, gx, 0);
#else
  __builtin_amdgcn_tensor_load_to_lds(g0, g1, g2, g3, 0);
#endif
#endif
}

// ---------------------------------------------------------------------------
// fp32 -> bf16 tensor conversion (once per launch per tensor)
// ---------------------------------------------------------------------------
__global__ __launch_bounds__(256) void cvt_bf16(
    const float* __restrict__ in, __bf16* __restrict__ out, long n)
{
  long stride = (long)gridDim.x * 256;
  for (long i = (long)blockIdx.x * 256 + threadIdx.x; i < n; i += stride)
    out[i] = f2bf(in[i]);
}

// ---------------------------------------------------------------------------
// Embedding: x = tok_emb[idx] + sinusoidal positional encoding (fp32)
// ---------------------------------------------------------------------------
__global__ __launch_bounds__(256) void embed_kernel(
    const int* __restrict__ idx, const float* __restrict__ tok_emb,
    float* __restrict__ x)
{
  const int bt  = blockIdx.x;
  const int t   = bt % T_;
  const int tok = idx[bt];
  for (int e = threadIdx.x; e < E_; e += 256) {
    float ang = (float)t * __expf((float)(e & ~1) * (-9.210340372f / (float)E_));
    float pe  = (e & 1) ? __cosf(ang) : __sinf(ang);
    x[(size_t)bt * E_ + e] = tok_emb[(size_t)tok * E_ + e] + pe;
  }
}

// ---------------------------------------------------------------------------
// LayerNorm: fp32 in, bf16 out (feeds the WMMA GEMMs)
// ---------------------------------------------------------------------------
__global__ __launch_bounds__(256) void ln_kernel(
    const float* __restrict__ x, const float* __restrict__ wgt,
    const float* __restrict__ bias, __bf16* __restrict__ out)
{
  __shared__ float red[256];
  const int row = blockIdx.x, tid = threadIdx.x;
  const float* xr = x + (size_t)row * E_;

  float s = 0.f;
  for (int i = tid; i < E_; i += 256) s += xr[i];
  red[tid] = s; __syncthreads();
  for (int st = 128; st > 0; st >>= 1) {
    if (tid < st) red[tid] += red[tid + st];
    __syncthreads();
  }
  float mu = red[0] / (float)E_;
  __syncthreads();

  float v = 0.f;
  for (int i = tid; i < E_; i += 256) { float d = xr[i] - mu; v += d * d; }
  red[tid] = v; __syncthreads();
  for (int st = 128; st > 0; st >>= 1) {
    if (tid < st) red[tid] += red[tid + st];
    __syncthreads();
  }
  float inv = rsqrtf(red[0] / (float)E_ + 1e-5f);

  for (int i = tid; i < E_; i += 256)
    out[(size_t)row * E_ + i] = f2bf((xr[i] - mu) * inv * wgt[i] + bias[i]);
}

// ---------------------------------------------------------------------------
// GEMM: C[M,N] = A[M,K]@B[K,N] (+bias)(+resid)(+relu), A/B bf16, f32 acc.
// 256 thr / 8 waves, 128x128 tile; wave = 32x64 = 2x4 WMMA tiles.
// TDM path: each wave DMAs its slice (A:16x32, B:4x128) into double-buffered
// LDS; sync via s_wait_tensorcnt + barrier; compute overlaps next-tile DMA.
// ---------------------------------------------------------------------------
static __device__ __forceinline__ void mma_step(
    const __bf16 (*Asr)[32], const __bf16 (*Bsr)[128],
    v8f acc[2][4], int lane, int wm, int wn)
{
  v16bf af[2], bfm[4];
#pragma unroll
  for (int mt = 0; mt < 2; ++mt) {
    int mr  = wm * 32 + mt * 16 + (lane & 15);
    int off = (lane >> 4) * 8;
    af[mt] = ldA(&Asr[mr][off], &Asr[mr][off + 16]);
  }
#pragma unroll
  for (int nt = 0; nt < 4; ++nt)          // B frag: n = lane%16, k in A table order
    for (int i = 0; i < 16; ++i)
      bfm[nt][i] = Bsr[(lane >> 4) * 8 + (i & 7) + (i >> 3) * 16][wn * 64 + nt * 16 + (lane & 15)];
#pragma unroll
  for (int mt = 0; mt < 2; ++mt)
#pragma unroll
    for (int nt = 0; nt < 4; ++nt)
      acc[mt][nt] = __builtin_amdgcn_wmma_f32_16x16x32_bf16(
          false, af[mt], false, bfm[nt], (short)0, acc[mt][nt], false, false);
}

__global__ __launch_bounds__(256) void gemm_wmma(
    const __bf16* __restrict__ A, const __bf16* __restrict__ Bw,
    const float* __restrict__ bias, const float* __restrict__ resid,
    void* __restrict__ C, int M, int N, int K, int relu, int out_bf16)
{
  __shared__ __align__(16) __bf16 As[2][128][32];
  __shared__ __align__(16) __bf16 Bs[2][32][128];

  const int tid  = threadIdx.x;
  const int lane = tid & 31;
  const int w    = tid >> 5;
  const int wm   = w & 3;            // 4 M-subblocks of 32 rows
  const int wn   = w >> 2;           // 2 N-subblocks of 64 cols
  const int m0   = blockIdx.y * 128;
  const int n0   = blockIdx.x * 128;

  v8f acc[2][4] = {};

#if HAVE_TDM
  auto issue_tiles = [&](int buf, int k0) {
    tdm_load_2d((unsigned)(size_t)&As[buf][w * 16][0],
                A + (size_t)(m0 + w * 16) * K + k0, 32, 16, K);
    tdm_load_2d((unsigned)(size_t)&Bs[buf][w * 4][0],
                Bw + (size_t)(k0 + w * 4) * N + n0, 128, 4, N);
  };
  int cur = 0;
  issue_tiles(0, 0);
  for (int k0 = 0; k0 < K; k0 += 32) {
    __builtin_amdgcn_s_wait_tensorcnt(0);    // this wave's tiles landed
    __syncthreads();                         // everyone's tiles visible
    if (k0 + 32 < K) issue_tiles(cur ^ 1, k0 + 32);
    mma_step(As[cur], Bs[cur], acc, lane, wm, wn);
    cur ^= 1;
  }
#else
  for (int k0 = 0; k0 < K; k0 += 32) {
    __syncthreads();
#pragma unroll
    for (int p = 0; p < 2; ++p) {            // A: 512 x 16B chunks
      int c = tid + p * 256;
      int r = c >> 2, off = (c & 3) * 8;
      *(v8bf*)&As[0][r][off] = *(const v8bf*)&A[(size_t)(m0 + r) * K + k0 + off];
    }
#pragma unroll
    for (int p = 0; p < 2; ++p) {            // B: 512 x 16B chunks
      int c = tid + p * 256;
      int r = c >> 4, off = (c & 15) * 8;
      *(v8bf*)&Bs[0][r][off] = *(const v8bf*)&Bw[(size_t)(k0 + r) * N + n0 + off];
    }
    __syncthreads();
    mma_step(As[0], Bs[0], acc, lane, wm, wn);
  }
#endif

  // Epilogue per C/D layout: m = 8*(lane>>4)+r, n = lane&15
#pragma unroll
  for (int mt = 0; mt < 2; ++mt)
#pragma unroll
    for (int nt = 0; nt < 4; ++nt)
#pragma unroll
      for (int r = 0; r < 8; ++r) {
        int m = m0 + wm * 32 + mt * 16 + (lane >> 4) * 8 + r;
        int n = n0 + wn * 64 + nt * 16 + (lane & 15);
        float vv = acc[mt][nt][r];
        if (bias)  vv += bias[n];
        if (resid) vv += resid[(size_t)m * N + n];
        if (relu)  vv = fmaxf(vv, 0.f);
        if (out_bf16) ((__bf16*)C)[(size_t)m * N + n] = f2bf(vv);
        else          ((float*)C)[(size_t)m * N + n] = vv;
      }
}

// ---------------------------------------------------------------------------
// Flash attention (bf16 in/out): 128 thr / 4 waves per 64 query rows of one
// (b,h). S = Q·Kt WMMA, online softmax, O += P·V WMMA.
// ---------------------------------------------------------------------------
__global__ __launch_bounds__(128) void attn_wmma(
    const __bf16* __restrict__ q, const __bf16* __restrict__ kbuf,
    const __bf16* __restrict__ vbuf, __bf16* __restrict__ o)
{
  __shared__ __align__(16) __bf16 Qs[64][64];      // [m][d]
  __shared__ __align__(16) __bf16 Kts[64][64];     // [d][s]  (transposed)
  __shared__ __align__(16) __bf16 Vs[64][64];      // [s][d]
  __shared__ __align__(16) __bf16 Ps[4][16][64];   // per-wave P tile [m][s]

  const int tid = threadIdx.x, lane = tid & 31, w = tid >> 5;
  const int mb = blockIdx.x;
  const int bh = blockIdx.y;
  const int b  = bh / H_, h = bh % H_;
  const int m0 = mb * 64;
  const float scale = 0.125f;                      // 1/sqrt(HS)

  const int dcol = tid & 63, rrow = tid >> 6;      // scalar loaders

  for (int p = 0; p < 32; ++p) {
    int mr = rrow + p * 2;
    Qs[mr][dcol] =
        f2bf(bf2f(q[((size_t)(b * T_ + m0 + mr)) * E_ + h * HS_ + dcol]) * scale);
  }
  __syncthreads();

  v16bf aq[2];
#pragma unroll
  for (int kk = 0; kk < 2; ++kk) {                 // HS=64 -> two K=32 steps
    int mr  = w * 16 + (lane & 15);
    int off = kk * 32 + (lane >> 4) * 8;
    aq[kk] = ldA(&Qs[mr][off], &Qs[mr][off + 16]);
  }

  v8f oacc[4] = {};
  float mrun[8], lrun[8];
#pragma unroll
  for (int r = 0; r < 8; ++r) { mrun[r] = -1e30f; lrun[r] = 0.f; }

  for (int kb = 0; kb <= mb; ++kb) {               // causal: keys <= queries
    int s0 = kb * 64;
    __syncthreads();
    for (int p = 0; p < 32; ++p) {                 // K transposed, scalar
      int sr = rrow + p * 2;
      Kts[dcol][sr] = kbuf[((size_t)(b * T_ + s0 + sr)) * E_ + h * HS_ + dcol];
    }
#pragma unroll
    for (int p = 0; p < 4; ++p) {                  // V, 16B chunks
      int c = tid + p * 128;
      int r = c >> 3, off = (c & 7) * 8;
      *(v8bf*)&Vs[r][off] =
          *(const v8bf*)&vbuf[((size_t)(b * T_ + s0 + r)) * E_ + h * HS_ + off];
    }
    __syncthreads();

    // S = Q @ Kt  (16x64 per wave)
    v8f sacc[4] = {};
#pragma unroll
    for (int kk = 0; kk < 2; ++kk)
#pragma unroll
      for (int nt = 0; nt < 4; ++nt) {
        v16bf bk;
        for (int i = 0; i < 16; ++i)
          bk[i] = Kts[kk * 32 + (lane >> 4) * 8 + (i & 7) + (i >> 3) * 16][nt * 16 + (lane & 15)];
        sacc[nt] = __builtin_amdgcn_wmma_f32_16x16x32_bf16(
            false, aq[kk], false, bk, (short)0, sacc[nt], false, false);
      }

    if (kb == mb) {                                // triangular mask on diag
#pragma unroll
      for (int nt = 0; nt < 4; ++nt)
#pragma unroll
        for (int r = 0; r < 8; ++r) {
          int nl = nt * 16 + (lane & 15);
          int ml = w * 16 + (lane >> 4) * 8 + r;
          if (nl > ml) sacc[nt][r] = -1e30f;
        }
    }

    // Online softmax: rows live in half-wave groups of 16 lanes
#pragma unroll
    for (int r = 0; r < 8; ++r) {
      float mnew = mrun[r];
#pragma unroll
      for (int nt = 0; nt < 4; ++nt) mnew = fmaxf(mnew, sacc[nt][r]);
#pragma unroll
      for (int d = 1; d < 16; d <<= 1)
        mnew = fmaxf(mnew, __shfl_xor(mnew, d, 32));
      float rsum = 0.f;
#pragma unroll
      for (int nt = 0; nt < 4; ++nt) {
        float pr = __expf(sacc[nt][r] - mnew);
        sacc[nt][r] = pr;
        rsum += pr;
      }
#pragma unroll
      for (int d = 1; d < 16; d <<= 1)
        rsum += __shfl_xor(rsum, d, 32);
      float alpha = __expf(mrun[r] - mnew);
      lrun[r] = lrun[r] * alpha + rsum;
      mrun[r] = mnew;
#pragma unroll
      for (int nt = 0; nt < 4; ++nt) oacc[nt][r] *= alpha;
    }

    // C-layout -> A-layout reshape through LDS
#pragma unroll
    for (int nt = 0; nt < 4; ++nt)
#pragma unroll
      for (int r = 0; r < 8; ++r)
        Ps[w][(lane >> 4) * 8 + r][nt * 16 + (lane & 15)] = f2bf(sacc[nt][r]);
    __syncthreads();

    // O += P @ V
#pragma unroll
    for (int kk = 0; kk < 2; ++kk) {
      int mr  = lane & 15;
      int off = kk * 32 + (lane >> 4) * 8;
      v16bf ap = ldA(&Ps[w][mr][off], &Ps[w][mr][off + 16]);
#pragma unroll
      for (int nt = 0; nt < 4; ++nt) {
        v16bf bv;
        for (int i = 0; i < 16; ++i)
          bv[i] = Vs[kk * 32 + (lane >> 4) * 8 + (i & 7) + (i >> 3) * 16][nt * 16 + (lane & 15)];
        oacc[nt] = __builtin_amdgcn_wmma_f32_16x16x32_bf16(
            false, ap, false, bv, (short)0, oacc[nt], false, false);
      }
    }
  }

#pragma unroll
  for (int nt = 0; nt < 4; ++nt)
#pragma unroll
    for (int r = 0; r < 8; ++r) {
      int mg = m0 + w * 16 + (lane >> 4) * 8 + r;
      int d  = nt * 16 + (lane & 15);
      o[((size_t)(b * T_ + mg)) * E_ + h * HS_ + d] = f2bf(oacc[nt][r] / lrun[r]);
    }
}

// ---------------------------------------------------------------------------
// Loss: per-row log-softmax + NLL, atomically accumulated
// ---------------------------------------------------------------------------
__global__ void zero_loss(float* loss) {
  if (threadIdx.x == 0 && blockIdx.x == 0) *loss = 0.f;
}

__global__ __launch_bounds__(256) void loss_kernel(
    const float* __restrict__ logits, const int* __restrict__ targets,
    float* __restrict__ loss)
{
  __shared__ float red[256];
  const int row = blockIdx.x, tid = threadIdx.x;
  const float* lr = logits + (size_t)row * V_;

  float mx = -1e30f;
  for (int i = tid; i < V_; i += 256) mx = fmaxf(mx, lr[i]);
  red[tid] = mx; __syncthreads();
  for (int st = 128; st > 0; st >>= 1) {
    if (tid < st) red[tid] = fmaxf(red[tid], red[tid + st]);
    __syncthreads();
  }
  mx = red[0]; __syncthreads();

  float s = 0.f;
  for (int i = tid; i < V_; i += 256) s += __expf(lr[i] - mx);
  red[tid] = s; __syncthreads();
  for (int st = 128; st > 0; st >>= 1) {
    if (tid < st) red[tid] += red[tid + st];
    __syncthreads();
  }
  if (tid == 0) {
    float lse = mx + __logf(red[0]);
    float lt  = lr[targets[row]];
    atomicAdd(loss, (lse - lt) / (float)BT_);
  }
}

// ---------------------------------------------------------------------------
extern "C" void kernel_launch(void* const* d_in, const int* in_sizes, int n_in,
                              void* d_out, int out_size, void* d_ws, size_t ws_size,
                              hipStream_t stream)
{
  const int*   idx     = (const int*)  d_in[0];
  const int*   targets = (const int*)  d_in[1];
  const float* tok_emb = (const float*)d_in[2];
  const float* wq   = (const float*)d_in[3];
  const float* wk   = (const float*)d_in[4];
  const float* wv   = (const float*)d_in[5];
  const float* wo   = (const float*)d_in[6];
  const float* bo   = (const float*)d_in[7];
  const float* ln1w = (const float*)d_in[8];
  const float* ln1b = (const float*)d_in[9];
  const float* w1   = (const float*)d_in[10];
  const float* b1   = (const float*)d_in[11];
  const float* w2   = (const float*)d_in[12];
  const float* b2   = (const float*)d_in[13];
  const float* ln2w = (const float*)d_in[14];
  const float* ln2b = (const float*)d_in[15];
  const float* lnfw = (const float*)d_in[16];
  const float* lnfb = (const float*)d_in[17];
  const float* lmw  = (const float*)d_in[18];
  const float* lmb  = (const float*)d_in[19];

  const size_t LEE = (size_t)L_ * E_ * E_;      // 8.4M elems
  const size_t LEF = (size_t)L_ * E_ * FF_;     // 33.5M elems
  const size_t EV  = (size_t)E_ * V_;           // 32.8M elems

  char* p = (char*)d_ws;
  auto alloc = [&](size_t bytes) { char* r = p; p += (bytes + 255) & ~(size_t)255; return r; };

  __bf16* bwq = (__bf16*)alloc(LEE * 2);
  __bf16* bwk = (__bf16*)alloc(LEE * 2);
  __bf16* bwv = (__bf16*)alloc(LEE * 2);
  __bf16* bwo = (__bf16*)alloc(LEE * 2);
  __bf16* bw1 = (__bf16*)alloc(LEF * 2);
  __bf16* bw2 = (__bf16*)alloc(LEF * 2);
  __bf16* blm = (__bf16*)alloc(EV  * 2);
  float*  x   = (float*) alloc((size_t)BT_ * E_ * 4);
  __bf16* hb  = (__bf16*)alloc((size_t)BT_ * E_ * 2);
  __bf16* qb  = (__bf16*)alloc((size_t)BT_ * E_ * 2);
  __bf16* kb  = (__bf16*)alloc((size_t)BT_ * E_ * 2);
  __bf16* vb  = (__bf16*)alloc((size_t)BT_ * E_ * 2);
  __bf16* ob  = (__bf16*)alloc((size_t)BT_ * E_ * 2);
  __bf16* fh  = (__bf16*)alloc((size_t)BT_ * FF_ * 2);

  float* logits   = (float*)d_out;
  float* loss_ptr = logits + (size_t)BT_ * V_;

  // One-time (per launch) weight down-conversion to bf16
  cvt_bf16<<<8192, 256, 0, stream>>>(wq,  bwq, (long)LEE);
  cvt_bf16<<<8192, 256, 0, stream>>>(wk,  bwk, (long)LEE);
  cvt_bf16<<<8192, 256, 0, stream>>>(wv,  bwv, (long)LEE);
  cvt_bf16<<<8192, 256, 0, stream>>>(wo,  bwo, (long)LEE);
  cvt_bf16<<<8192, 256, 0, stream>>>(w1,  bw1, (long)LEF);
  cvt_bf16<<<8192, 256, 0, stream>>>(w2,  bw2, (long)LEF);
  cvt_bf16<<<8192, 256, 0, stream>>>(lmw, blm, (long)EV);

  embed_kernel<<<BT_, 256, 0, stream>>>(idx, tok_emb, x);

  dim3 gE(E_ / 128, BT_ / 128);     // N=1024
  dim3 gF(FF_ / 128, BT_ / 128);    // N=4096
  dim3 gV(V_ / 128, BT_ / 128);     // N=32000 = 250*128

  for (int l = 0; l < L_; ++l) {
    const __bf16* lwq = bwq + (size_t)l * E_ * E_;
    const __bf16* lwk = bwk + (size_t)l * E_ * E_;
    const __bf16* lwv = bwv + (size_t)l * E_ * E_;
    const __bf16* lwo = bwo + (size_t)l * E_ * E_;
    const __bf16* lw1 = bw1 + (size_t)l * E_ * FF_;
    const __bf16* lw2 = bw2 + (size_t)l * FF_ * E_;

    ln_kernel<<<BT_, 256, 0, stream>>>(x, ln1w + l * E_, ln1b + l * E_, hb);
    gemm_wmma<<<gE, 256, 0, stream>>>(hb, lwq, nullptr, nullptr, qb, BT_, E_, E_, 0, 1);
    gemm_wmma<<<gE, 256, 0, stream>>>(hb, lwk, nullptr, nullptr, kb, BT_, E_, E_, 0, 1);
    gemm_wmma<<<gE, 256, 0, stream>>>(hb, lwv, nullptr, nullptr, vb, BT_, E_, E_, 0, 1);
    attn_wmma<<<dim3(T_ / 64, B_ * H_), 128, 0, stream>>>(qb, kb, vb, ob);
    gemm_wmma<<<gE, 256, 0, stream>>>(ob, lwo, bo + l * E_, x, x, BT_, E_, E_, 0, 0);
    ln_kernel<<<BT_, 256, 0, stream>>>(x, ln2w + l * E_, ln2b + l * E_, hb);
    gemm_wmma<<<gF, 256, 0, stream>>>(hb, lw1, b1 + l * FF_, nullptr, fh, BT_, FF_, E_, 1, 1);
    gemm_wmma<<<gE, 256, 0, stream>>>(fh, lw2, b2 + l * E_, x, x, BT_, E_, FF_, 0, 0);
  }

  ln_kernel<<<BT_, 256, 0, stream>>>(x, lnfw, lnfb, hb);
  gemm_wmma<<<gV, 256, 0, stream>>>(hb, blm, lmb, nullptr, logits, BT_, V_, E_, 0, 0);
  zero_loss<<<1, 1, 0, stream>>>(loss_ptr);
  loss_kernel<<<BT_, 256, 0, stream>>>(logits, targets, loss_ptr);
}
